// CausalSelfAttention_32530082300503
// MI455X (gfx1250) — matrix operand order repaired and checked by reading, not compiled, not run on hardware
//
#include <hip/hip_runtime.h>
#include <hip/hip_bf16.h>

// ---------------------------------------------------------------------------
// Causal self-attention for MI455X (gfx1250, wave32, WMMA).
// ~120 GFLOP vs ~76MB of I/O @ 23.3 TB/s -> strongly compute-bound -> bf16
// WMMA path (v_wmma_f32_16x16x32_bf16, fp32 accumulate).  All bf16
// intermediates (~102MB) are L2-resident (192MB L2): fragments are loaded
// straight from global with b128 loads in contiguous k order (k = 16*(lane/16) + i, A and B alike).
// Each wave computes a 32x64 tile (8 accumulators).  The k-loop is an
// explicit ping-pong pipeline over 64-wide k-steps with unrolling DISABLED:
// the compiler must not deepen the pipeline, which would re-introduce
// register rotation copies (v_mov + WMMA->VALU hazard NOPs) and push VGPR
// usage past 256 (s_set_vgpr_msb addressing + occupancy loss).
// V is produced transposed so the PV GEMM also gets contiguous b128 loads.
// ---------------------------------------------------------------------------

typedef __bf16 bf16_t;
typedef __attribute__((ext_vector_type(16))) __bf16 v16bf;
typedef __attribute__((ext_vector_type(8)))  float  v8f;

#define B_  4
#define S_  2048
#define D_  1024
#define M_  (B_ * S_)        // 8192 total tokens
#define NX_ (M_ * D_)        // 8388608 elements of X
#define NW_ (D_ * D_)        // 1048576 elements per weight

__device__ __forceinline__ v8f wmma_bf16(v16bf a, v16bf b, v8f c) {
  return __builtin_amdgcn_wmma_f32_16x16x32_bf16(
      /*neg_a=*/false, a, /*neg_b=*/false, b,
      /*c_mod=*/(short)0, c, /*reuse_a=*/false, /*reuse_b=*/false);
}

// A fragment (16x32 bf16, MxK) from a row-major [rows, ld] matrix.
// Lane L: M = mbase + L%16;  K = kbase + 16*(L/16) + i
// -> two contiguous 16B loads per lane.
__device__ __forceinline__ v16bf load_frag_A(const bf16_t* __restrict__ base,
                                             int ld, int mbase, int kbase, int lane) {
  const bf16_t* p = base + (size_t)(mbase + (lane & 15)) * ld
                         + kbase + ((lane >> 4) << 4);
  union { uint4 q[2]; v16bf v; } u;
  u.q[0] = *(const uint4*)(p);
  u.q[1] = *(const uint4*)(p + 8);
  return u.v;
}

// B fragment (32x16 bf16, KxN) where element (k,n) = W[n][k], W row-major
// [N, ld].  Lane L: N = nbase + L%16; K = kbase + 16*(L/16) + i
// -> 16 contiguous bf16 = two 16B loads per lane.
__device__ __forceinline__ v16bf load_frag_BT(const bf16_t* __restrict__ base,
                                              int ld, int nbase, int kbase, int lane) {
  const bf16_t* p = base + (size_t)(nbase + (lane & 15)) * ld
                         + kbase + ((lane >> 4) << 4);
  union { uint4 q[2]; v16bf v; } u;
  u.q[0] = *(const uint4*)(p);
  u.q[1] = *(const uint4*)(p + 8);
  return u.v;
}

// 8 independent WMMAs: 32x64 tile += (32x32) x (32x64)
__device__ __forceinline__ void wmma_block(v8f acc[2][4],
    const v16bf& x0, const v16bf& x1,
    const v16bf& y0, const v16bf& y1, const v16bf& y2, const v16bf& y3) {
  acc[0][0] = wmma_bf16(x0, y0, acc[0][0]);
  acc[0][1] = wmma_bf16(x0, y1, acc[0][1]);
  acc[0][2] = wmma_bf16(x0, y2, acc[0][2]);
  acc[0][3] = wmma_bf16(x0, y3, acc[0][3]);
  acc[1][0] = wmma_bf16(x1, y0, acc[1][0]);
  acc[1][1] = wmma_bf16(x1, y1, acc[1][1]);
  acc[1][2] = wmma_bf16(x1, y2, acc[1][2]);
  acc[1][3] = wmma_bf16(x1, y3, acc[1][3]);
}

// Ping-pong pipelined 32x64 GEMM tile over K = kend (multiple of 64, >= 64).
// A: row-major [*, lda] at row mbase; B: "transposed" operand, row-major
// [*, ldb] at row nbase (element (k,n) = B[n][k]).
__device__ __forceinline__ void mm_pipe(
    const bf16_t* __restrict__ A, int lda, int mbase,
    const bf16_t* __restrict__ B, int ldb, int nbase,
    int kend, int lane, v8f acc[2][4]) {
  // buffer A <- k-step 0
  v16bf aA0 = load_frag_A(A, lda, mbase,      0, lane);
  v16bf aA1 = load_frag_A(A, lda, mbase + 16, 0, lane);
  v16bf bA0 = load_frag_BT(B, ldb, nbase +  0, 0, lane);
  v16bf bA1 = load_frag_BT(B, ldb, nbase + 16, 0, lane);
  v16bf bA2 = load_frag_BT(B, ldb, nbase + 32, 0, lane);
  v16bf bA3 = load_frag_BT(B, ldb, nbase + 48, 0, lane);
  // buffer B <- k-step 32
  v16bf aB0 = load_frag_A(A, lda, mbase,      32, lane);
  v16bf aB1 = load_frag_A(A, lda, mbase + 16, 32, lane);
  v16bf bB0 = load_frag_BT(B, ldb, nbase +  0, 32, lane);
  v16bf bB1 = load_frag_BT(B, ldb, nbase + 16, 32, lane);
  v16bf bB2 = load_frag_BT(B, ldb, nbase + 32, 32, lane);
  v16bf bB3 = load_frag_BT(B, ldb, nbase + 48, 32, lane);

#pragma clang loop unroll(disable)
  for (int k = 64; k < kend; k += 64) {
    // consume buffer A (data for k-64), refill it with k
    wmma_block(acc, aA0, aA1, bA0, bA1, bA2, bA3);
    aA0 = load_frag_A(A, lda, mbase,      k, lane);
    aA1 = load_frag_A(A, lda, mbase + 16, k, lane);
    bA0 = load_frag_BT(B, ldb, nbase +  0, k, lane);
    bA1 = load_frag_BT(B, ldb, nbase + 16, k, lane);
    bA2 = load_frag_BT(B, ldb, nbase + 32, k, lane);
    bA3 = load_frag_BT(B, ldb, nbase + 48, k, lane);
    // consume buffer B (data for k-32), refill it with k+32
    wmma_block(acc, aB0, aB1, bB0, bB1, bB2, bB3);
    aB0 = load_frag_A(A, lda, mbase,      k + 32, lane);
    aB1 = load_frag_A(A, lda, mbase + 16, k + 32, lane);
    bB0 = load_frag_BT(B, ldb, nbase +  0, k + 32, lane);
    bB1 = load_frag_BT(B, ldb, nbase + 16, k + 32, lane);
    bB2 = load_frag_BT(B, ldb, nbase + 32, k + 32, lane);
    bB3 = load_frag_BT(B, ldb, nbase + 48, k + 32, lane);
  }
  wmma_block(acc, aA0, aA1, bA0, bA1, bA2, bA3);
  wmma_block(acc, aB0, aB1, bB0, bB1, bB2, bB3);
}

// ---------------------------------------------------------------------------
// Kernel 1: fp32 -> bf16 conversion of X, Wq, Wk, Wv.
// ---------------------------------------------------------------------------
__global__ __launch_bounds__(256) void csa_cvt_bf16(
    const float* __restrict__ X,  const float* __restrict__ Wq,
    const float* __restrict__ Wk, const float* __restrict__ Wv,
    bf16_t* __restrict__ Xb, bf16_t* __restrict__ Wqb,
    bf16_t* __restrict__ Wkb, bf16_t* __restrict__ Wvb) {
  const long total = (long)NX_ + 3L * NW_;
  for (long i = (long)blockIdx.x * blockDim.x + threadIdx.x; i < total;
       i += (long)gridDim.x * blockDim.x) {
    if (i < NX_) {
      Xb[i] = (bf16_t)X[i];
    } else {
      long j = i - NX_;
      if (j < NW_)            Wqb[j]          = (bf16_t)Wq[j];
      else if (j < 2L * NW_)  Wkb[j - NW_]    = (bf16_t)Wk[j - NW_];
      else                    Wvb[j - 2L*NW_] = (bf16_t)Wv[j - 2L*NW_];
    }
  }
}

// ---------------------------------------------------------------------------
// Kernel 2: C[m,n] = sum_k A[m,k] * W[n,k]   (X @ W^T), bf16 in, bf16 out.
// One wave -> 32x64 tile.  TOUT=true stores the result TRANSPOSED per batch
// (Vt[b][n][s]) so the PV GEMM can use contiguous B loads.
// ---------------------------------------------------------------------------
template <bool TOUT>
__global__ __launch_bounds__(128) void csa_qkv_gemm(
    const bf16_t* __restrict__ A, const bf16_t* __restrict__ W,
    bf16_t* __restrict__ C) {
  const int lane = threadIdx.x & 31;
  const int wave = blockIdx.x * (blockDim.x >> 5) + (threadIdx.x >> 5);
  const int mbase = (wave >> 4) * 32;    // 256 m-tiles of 32
  const int nbase = (wave & 15) * 64;    // 16 n-strips of 64

  v8f acc[2][4] = {};
  mm_pipe(A, D_, mbase, W, D_, nbase, D_, lane, acc);

  const int nlo = lane & 15;
  const int mhi = (lane >> 4) << 3;
#pragma unroll
  for (int i = 0; i < 2; ++i) {
#pragma unroll
    for (int t = 0; t < 4; ++t) {
#pragma unroll
      for (int r = 0; r < 8; ++r) {
        int m = mbase + i * 16 + r + mhi;      // global token index
        int n = nbase + t * 16 + nlo;
        if (!TOUT) {
          C[(size_t)m * D_ + n] = (bf16_t)acc[i][t][r];
        } else {
          int bb = m >> 11;                    // batch (tile never crosses)
          int s  = m & (S_ - 1);
          C[((size_t)bb * D_ + n) * S_ + s] = (bf16_t)acc[i][t][r];
        }
      }
    }
  }
}

// ---------------------------------------------------------------------------
// Kernel 3: causal scores  S[b,q,kv] = (sum_d Q[q,d]K[kv,d]) / 32, masked.
// One wave -> 32(q) x 64(kv) tile; strips fully above the diagonal skip the
// WMMA loop and just store -inf.  Diagonal strips compute all sub-tiles and
// mask at store (keeps the pipelined loop uniform, EXEC all-ones).
// ---------------------------------------------------------------------------
__global__ __launch_bounds__(128) void csa_scores(
    const bf16_t* __restrict__ Q, const bf16_t* __restrict__ K,
    bf16_t* __restrict__ Sc) {
  const int lane = threadIdx.x & 31;
  const int wave = blockIdx.x * (blockDim.x >> 5) + (threadIdx.x >> 5);
  const int b   = wave >> 11;          // 2048 waves per batch
  const int rem = wave & 2047;
  const int qt  = rem >> 5;            // 64 q-tiles of 32
  const int ks  = rem & 31;            // 32 kv-strips of 64

  const bf16_t* Qp = Q + (size_t)b * S_ * D_;
  const bf16_t* Kp = K + (size_t)b * S_ * D_;
  bf16_t*       Sp = Sc + (size_t)b * S_ * S_;

  const int qbase  = qt * 32;
  const int kvbase = ks * 64;

  v8f acc[2][4] = {};
  if (kvbase <= qbase + 31)            // wave-uniform: strip touches diagonal
    mm_pipe(Qp, D_, qbase, Kp, D_, kvbase, D_, lane, acc);

  const int nlo = lane & 15;
  const int mhi = (lane >> 4) << 3;
  const float kNegInf = -__builtin_inff();
#pragma unroll
  for (int i = 0; i < 2; ++i) {
#pragma unroll
    for (int t = 0; t < 4; ++t) {
      int kv = kvbase + t * 16 + nlo;
#pragma unroll
      for (int r = 0; r < 8; ++r) {
        int q = qbase + i * 16 + r + mhi;
        float s = (kv > q) ? kNegInf : acc[i][t][r] * 0.03125f;  // 1/sqrt(1024)
        Sp[(size_t)q * S_ + kv] = (bf16_t)s;
      }
    }
  }
}

// ---------------------------------------------------------------------------
// Kernel 4: in-place row softmax over bf16 scores.  One 256-thread block per
// row (2048 entries -> 8 per thread).  Masked -inf entries become 0.
// ---------------------------------------------------------------------------
__global__ __launch_bounds__(256) void csa_softmax(bf16_t* __restrict__ Sc) {
  const int row = blockIdx.x;                 // 0 .. B*S-1
  bf16_t* p = Sc + (size_t)row * S_;
  const int tid = threadIdx.x;

  float v[8];
  float mx = -__builtin_inff();
#pragma unroll
  for (int j = 0; j < 8; ++j) {
    float f = (float)p[tid + j * 256];
    v[j] = f;
    mx = fmaxf(mx, f);
  }

  __shared__ float red[256];
  red[tid] = mx;
  __syncthreads();
  for (int off = 128; off > 0; off >>= 1) {
    if (tid < off) red[tid] = fmaxf(red[tid], red[tid + off]);
    __syncthreads();
  }
  mx = red[0];
  __syncthreads();

  float sum = 0.0f;
#pragma unroll
  for (int j = 0; j < 8; ++j) {
    float e = __expf(v[j] - mx);   // exp(-inf - finite) == 0
    v[j] = e;
    sum += e;
  }
  red[tid] = sum;
  __syncthreads();
  for (int off = 128; off > 0; off >>= 1) {
    if (tid < off) red[tid] += red[tid + off];
    __syncthreads();
  }
  const float inv = 1.0f / red[0];
#pragma unroll
  for (int j = 0; j < 8; ++j)
    p[tid + j * 256] = (bf16_t)(v[j] * inv);
}

// ---------------------------------------------------------------------------
// Kernel 5: O[b,q,e] = sum_kv P[b,q,kv] * Vt[b,e,kv]  (fp32 out).
// V stored transposed per batch -> contiguous b128 B loads.  One wave ->
// 32(q) x 64(e) tile; kv range = causal limit rounded up to 64 (the extra
// P columns are exact zeros from softmax, and all addresses stay in-bounds).
// ---------------------------------------------------------------------------
__global__ __launch_bounds__(128) void csa_pv(
    const bf16_t* __restrict__ P, const bf16_t* __restrict__ Vt,
    float* __restrict__ O) {
  const int lane = threadIdx.x & 31;
  const int wave = blockIdx.x * (blockDim.x >> 5) + (threadIdx.x >> 5);
  const int b   = wave >> 10;          // 1024 waves per batch
  const int rem = wave & 1023;
  const int qt  = rem >> 4;            // 64 q-tiles of 32
  const int ns  = rem & 15;            // 16 e-strips of 64

  const bf16_t* Pp  = P  + (size_t)b * S_ * S_;
  const bf16_t* Vtp = Vt + (size_t)b * D_ * S_;   // [D_, S_] row-major

  const int qbase = qt * 32;
  const int nbase = ns * 64;
  const int kend  = (qbase + 32 + 63) & ~63;      // causal limit, 64-aligned

  v8f acc[2][4] = {};
  mm_pipe(Pp, S_, qbase, Vtp, S_, nbase, kend, lane, acc);

  const int nlo = lane & 15;
  const int mhi = (lane >> 4) << 3;
#pragma unroll
  for (int i = 0; i < 2; ++i) {
#pragma unroll
    for (int t = 0; t < 4; ++t) {
#pragma unroll
      for (int r = 0; r < 8; ++r) {
        int q = qbase + i * 16 + r + mhi;
        O[((size_t)b * S_ + q) * D_ + nbase + t * 16 + nlo] = acc[i][t][r];
      }
    }
  }
}

// ---------------------------------------------------------------------------
extern "C" void kernel_launch(void* const* d_in, const int* in_sizes, int n_in,
                              void* d_out, int out_size, void* d_ws, size_t ws_size,
                              hipStream_t stream) {
  const float* X  = (const float*)d_in[0];
  const float* Wq = (const float*)d_in[1];
  const float* Wk = (const float*)d_in[2];
  const float* Wv = (const float*)d_in[3];
  float* Out = (float*)d_out;

  // Workspace layout (bf16):
  //   Xb [M_,D_] | Wqb,Wkb,Wvb [D_,D_] | Qb,Kb [M_,D_] | Vt [B_,D_,S_] | Sc [B_,S_,S_]
  char* ws = (char*)d_ws;
  bf16_t* Xb  = (bf16_t*)ws;                  ws += (size_t)NX_ * 2;
  bf16_t* Wqb = (bf16_t*)ws;                  ws += (size_t)NW_ * 2;
  bf16_t* Wkb = (bf16_t*)ws;                  ws += (size_t)NW_ * 2;
  bf16_t* Wvb = (bf16_t*)ws;                  ws += (size_t)NW_ * 2;
  bf16_t* Qb  = (bf16_t*)ws;                  ws += (size_t)NX_ * 2;
  bf16_t* Kb  = (bf16_t*)ws;                  ws += (size_t)NX_ * 2;
  bf16_t* Vt  = (bf16_t*)ws;                  ws += (size_t)NX_ * 2;
  bf16_t* Sc  = (bf16_t*)ws;                  // B_*S_*S_ bf16

  // 1) fp32 -> bf16
  csa_cvt_bf16<<<4096, 256, 0, stream>>>(X, Wq, Wk, Wv, Xb, Wqb, Wkb, Wvb);

  // 2) Q/K/V projections: 4096 waves each -> 1024 blocks of 128 threads.
  //    V is stored transposed per batch.
  csa_qkv_gemm<false><<<1024, 128, 0, stream>>>(Xb, Wqb, Qb);
  csa_qkv_gemm<false><<<1024, 128, 0, stream>>>(Xb, Wkb, Kb);
  csa_qkv_gemm<true ><<<1024, 128, 0, stream>>>(Xb, Wvb, Vt);

  // 3) causal scores: 8192 waves -> 2048 blocks
  csa_scores<<<2048, 128, 0, stream>>>(Qb, Kb, Sc);

  // 4) row softmax: one block per row
  csa_softmax<<<B_ * S_, 256, 0, stream>>>(Sc);

  // 5) P @ V: 4096 waves -> 1024 blocks
  csa_pv<<<1024, 128, 0, stream>>>(Sc, Vt, Out);
}
